// BboxEncoder_6511170421184
// MI455X (gfx1250) — matrix operand order repaired and checked by reading, not compiled, not run on hardware
//
#include <hip/hip_runtime.h>
#include <math.h>

typedef unsigned short u16;
typedef unsigned int   u32;
typedef unsigned long long u64;
typedef __attribute__((ext_vector_type(16))) __bf16 v16bf;
typedef __attribute__((ext_vector_type(8)))  float  v8f;
typedef __attribute__((ext_vector_type(8)))  int    v8i;
typedef int v4i_t __attribute__((vector_size(16)));

#ifndef __has_builtin
#define __has_builtin(x) 0
#endif
#if __has_builtin(__builtin_amdgcn_global_load_async_to_lds_b128)
#define HAVE_ASYNC_LDS 1
#else
#define HAVE_ASYNC_LDS 0
#endif

typedef __attribute__((address_space(1))) v4i_t as1_v4i;
typedef __attribute__((address_space(3))) v4i_t as3_v4i;

// ---------- helpers ----------
static __device__ __forceinline__ u16 f2bf(float f) {
  u32 u = __builtin_bit_cast(u32, f);
  u32 r = (u + 0x7FFFu + ((u >> 16) & 1u)) >> 16;   // round-to-nearest-even
  return (u16)r;
}
static __device__ __forceinline__ float bf2f(u32 lo16) {
  return __builtin_bit_cast(float, lo16 << 16);
}
static __device__ __forceinline__ v16bf as_bf(v8i v) {
  return __builtin_bit_cast(v16bf, v);
}
static __device__ __forceinline__ v8i pack8(int4 lo, int4 hi) {
  v8i r; r[0]=lo.x; r[1]=lo.y; r[2]=lo.z; r[3]=lo.w;
         r[4]=hi.x; r[5]=hi.y; r[6]=hi.z; r[7]=hi.w;
  return r;
}
static __device__ __forceinline__ v8f zero8() {
  return (v8f){0.f,0.f,0.f,0.f,0.f,0.f,0.f,0.f};
}
static __device__ __forceinline__ float redmax16(float x) {
  #pragma unroll
  for (int o = 8; o > 0; o >>= 1) x = fmaxf(x, __shfl_xor(x, o, 32));
  return x;
}
static __device__ __forceinline__ float redsum16(float x) {
  #pragma unroll
  for (int o = 8; o > 0; o >>= 1) x += __shfl_xor(x, o, 32);
  return x;
}

#if HAVE_ASYNC_LDS
static __device__ __forceinline__ void async_b128(const void* g, void* l) {
  __builtin_amdgcn_global_load_async_to_lds_b128(
      (as1_v4i*)(u64)g, (as3_v4i*)(u32)(u64)l, 0, 0);
}
#endif
static __device__ __forceinline__ void wait_async0() {
#if __has_builtin(__builtin_amdgcn_s_wait_asynccnt)
  __builtin_amdgcn_s_wait_asynccnt(0);
#else
  asm volatile("s_wait_asynccnt 0" ::: "memory");
#endif
}

// ---------- weight transpose + f32 -> bf16 convert ----------
__global__ __launch_bounds__(256) void transpose_bf16(
    const float* __restrict__ src, u16* __restrict__ dst, int K, int N) {
  __shared__ float tile[32][33];
  size_t zoff = (size_t)blockIdx.z * (size_t)K * (size_t)N;
  src += zoff; dst += zoff;
  int n0 = blockIdx.x * 32, k0 = blockIdx.y * 32;
  int tx = threadIdx.x, ty = threadIdx.y;          // 32 x 8
  #pragma unroll
  for (int i = 0; i < 32; i += 8)
    tile[ty + i][tx] = src[(size_t)(k0 + ty + i) * N + (n0 + tx)];
  __syncthreads();
  #pragma unroll
  for (int i = 0; i < 32; i += 8)
    dst[(size_t)(n0 + ty + i) * K + (k0 + tx)] = f2bf(tile[tx][ty + i]);
}

// ---------- embedding stage 1 ----------
__global__ __launch_bounds__(256) void embed1_kernel(
    const float* __restrict__ x, const float* __restrict__ W,
    const float* __restrict__ bias, u16* __restrict__ out) {
  int m = blockIdx.y;
  int e = blockIdx.x * 256 + threadIdx.x;
  const float* xr = x + (size_t)m * 4;
  float acc = bias[e];
  #pragma unroll
  for (int d = 0; d < 4; d++) acc += xr[d] * W[d * 1024 + e];
  out[(size_t)m * 1024 + e] = f2bf(acc);
}

// ---------- LayerNorm (f32 in) -> bf16 out ----------
__global__ __launch_bounds__(256) void ln_bf16(
    const float* __restrict__ h, const float* __restrict__ g,
    const float* __restrict__ b, u16* __restrict__ out) {
  int row = blockIdx.x, tid = threadIdx.x;
  const float* x = h + (size_t)row * 1024;
  float v[4], s1 = 0.f, s2 = 0.f;
  #pragma unroll
  for (int i = 0; i < 4; i++) {
    float t = x[tid + i * 256];
    v[i] = t; s1 += t; s2 += t * t;
  }
  #pragma unroll
  for (int o = 16; o > 0; o >>= 1) {
    s1 += __shfl_xor(s1, o, 32);
    s2 += __shfl_xor(s2, o, 32);
  }
  __shared__ float r1[8], r2[8];
  int w = tid >> 5, lane = tid & 31;
  if (lane == 0) { r1[w] = s1; r2[w] = s2; }
  __syncthreads();
  s1 = r1[lane & 7]; s2 = r2[lane & 7];
  #pragma unroll
  for (int o = 4; o > 0; o >>= 1) {
    s1 += __shfl_xor(s1, o, 32);
    s2 += __shfl_xor(s2, o, 32);
  }
  float mean = s1 * (1.0f / 1024.0f);
  float var  = s2 * (1.0f / 1024.0f) - mean * mean;
  float rstd = rsqrtf(var + 1e-5f);
  #pragma unroll
  for (int i = 0; i < 4; i++) {
    int c = tid + i * 256;
    out[(size_t)row * 1024 + c] = f2bf((v[i] - mean) * rstd * g[c] + b[c]);
  }
}

// ---------- bf16 WMMA GEMM, 128x128 tile, async double-buffered LDS --------
// MODE: 0 bf16+bias | 1 f32+bias | 2 f32+bias+res | 3 bf16+bias+GELU | 4 f32
template <int MODE>
__global__ __launch_bounds__(256) void gemm_bf16(
    const u16* __restrict__ A, const u16* __restrict__ Wt,
    const float* __restrict__ bias, const float* res,
    float* outF, u16* outB, int M, int N, int K) {
  __shared__ int lA[2][128 * 20];
  __shared__ int lB[2][128 * 20];

  int tid = threadIdx.x;
  int wid = tid >> 5, lane = tid & 31;
  int hh = lane >> 4, nl = lane & 15;
  int mBase = blockIdx.x * 128, nBase = blockIdx.y * 128;
  int wm = (wid >> 1) * 32;
  int wn = (wid & 1) * 64;

  v8f acc[2][4];
  #pragma unroll
  for (int f = 0; f < 2; f++)
    #pragma unroll
    for (int g = 0; g < 4; g++) acc[f][g] = zero8();

  int lrow = tid >> 1;
  int lseg = (tid & 1) * 8;
  const int nk = K >> 5;

#if HAVE_ASYNC_LDS
  auto issue = [&](int kt, int buf) {
    const u16* ap = A  + (size_t)(mBase + lrow) * K + kt * 32 + lseg * 2;
    const u16* bp = Wt + (size_t)(nBase + lrow) * K + kt * 32 + lseg * 2;
    int* la = &lA[buf][lrow * 20 + lseg];
    int* lb = &lB[buf][lrow * 20 + lseg];
    async_b128(ap, la);     async_b128(ap + 8, la + 4);
    async_b128(bp, lb);     async_b128(bp + 8, lb + 4);
  };
  issue(0, 0);
  int cur = 0;
#endif

  for (int kt = 0; kt < nk; kt++) {
    const int* bA;
    const int* bB;
#if HAVE_ASYNC_LDS
    wait_async0();
    __syncthreads();
    if (kt + 1 < nk) issue(kt + 1, cur ^ 1);
    bA = lA[cur]; bB = lB[cur];
#else
    {
      const u16* ap = A  + (size_t)(mBase + lrow) * K + kt * 32 + lseg * 2;
      const u16* bp = Wt + (size_t)(nBase + lrow) * K + kt * 32 + lseg * 2;
      int4 a0 = *(const int4*)ap, a1 = *(const int4*)(ap + 8);
      int4 b0 = *(const int4*)bp, b1 = *(const int4*)(bp + 8);
      __syncthreads();
      *(int4*)(&lA[0][lrow * 20 + lseg])     = a0;
      *(int4*)(&lA[0][lrow * 20 + lseg + 4]) = a1;
      *(int4*)(&lB[0][lrow * 20 + lseg])     = b0;
      *(int4*)(&lB[0][lrow * 20 + lseg + 4]) = b1;
      __syncthreads();
      bA = lA[0]; bB = lB[0];
    }
#endif
    v8i bfrag[4];
    #pragma unroll
    for (int g = 0; g < 4; g++) {
      const int* br = bB + (wn + g * 16 + nl) * 20 + 8 * hh;
      bfrag[g] = pack8(*(const int4*)br, *(const int4*)(br + 4));
    }
    #pragma unroll
    for (int f = 0; f < 2; f++) {
      const int* ar = bA + (wm + f * 16 + nl) * 20;
      v16bf abf = as_bf(pack8(*(const int4*)(ar + 8 * hh),
                              *(const int4*)(ar + 8 * hh + 4)));
      #pragma unroll
      for (int g = 0; g < 4; g++)
        acc[f][g] = __builtin_amdgcn_wmma_f32_16x16x32_bf16(
            false, abf, false, as_bf(bfrag[g]), (short)0, acc[f][g],
            false, false);
    }
#if HAVE_ASYNC_LDS
    cur ^= 1;
#endif
  }

  #pragma unroll
  for (int f = 0; f < 2; f++) {
    int row0 = mBase + wm + f * 16 + hh * 8;
    #pragma unroll
    for (int g = 0; g < 4; g++) {
      int col = nBase + wn + g * 16 + nl;
      float bv = (MODE == 4) ? 0.0f : bias[col];
      #pragma unroll
      for (int j = 0; j < 8; j++) {
        size_t idx = (size_t)(row0 + j) * N + col;
        float vv = acc[f][g][j] + bv;
        if (MODE == 0) {
          outB[idx] = f2bf(vv);
        } else if (MODE == 1) {
          outF[idx] = vv;
        } else if (MODE == 2) {
          outF[idx] = vv + res[idx];
        } else if (MODE == 3) {
          outB[idx] = f2bf(0.5f * vv * (1.0f + erff(vv * 0.70710678118654752f)));
        } else {
          outF[idx] = vv;
        }
      }
    }
  }
}

// ---------- WMMA flash attention -------------------------------------------
// grid = B*H*(T/128) blocks; 8 waves/block; wave owns 16-query tile.
// Q,K,V,Y bf16 [B*T][1024], head hd at column hd*64. scale = 1/8. Causal.
// Per-wave LDS only; relies on in-wave DS ordering (no barriers).
__global__ __launch_bounds__(256) void attn_wmma(
    const u16* __restrict__ Q, const u16* __restrict__ Kp,
    const u16* __restrict__ Vp, u16* __restrict__ Y) {
  __shared__ int pP[8 * 16 * 20];    // per wave: P tile 16x32 bf16 (stride 20dw)
  __shared__ int pV[8 * 64 * 20];    // per wave: V^T 64x32 bf16 (stride 20dw)

  int tid = threadIdx.x, w = tid >> 5, lane = tid & 31;
  int hh = lane >> 4, nl = lane & 15;
  int bh = blockIdx.x >> 2;          // (b*16 + hd)
  int qblk = blockIdx.x & 3;
  int b = bh >> 4, hd = bh & 15;
  int q0 = qblk * 128 + w * 16;
  size_t base = ((size_t)b * 512) * 1024 + hd * 64;

  int* pPw = pP + w * 320;
  int* pVw = pV + w * 1280;
  u16* pPu = (u16*)pPw;
  u16* pVu = (u16*)pVw;

  // Q A-fragments (kd halves 0..31, 32..63) straight from global
  const int* qrow = (const int*)(Q + base + (size_t)(q0 + nl) * 1024);
  v16bf qf0 = as_bf(pack8(*(const int4*)(qrow + 8 * hh),
                          *(const int4*)(qrow + 8 * hh + 4)));
  v16bf qf1 = as_bf(pack8(*(const int4*)(qrow + 16 + 8 * hh),
                          *(const int4*)(qrow + 16 + 8 * hh + 4)));

  v8f O[4];
  #pragma unroll
  for (int g = 0; g < 4; g++) O[g] = zero8();
  float m8[8], l8[8];
  #pragma unroll
  for (int j = 0; j < 8; j++) { m8[j] = -INFINITY; l8[j] = 0.f; }

  for (int kb = 0; kb < q0 + 16; kb += 32) {
    // ---- S = Q K^T for two 16-key tiles
    v8f s0 = zero8(), s1 = zero8();
    {
      const int* kr0 = (const int*)(Kp + base + (size_t)(kb + nl) * 1024);
      v16bf k00 = as_bf(pack8(*(const int4*)(kr0 + 8 * hh),
                              *(const int4*)(kr0 + 8 * hh + 4)));
      v16bf k01 = as_bf(pack8(*(const int4*)(kr0 + 16 + 8 * hh),
                              *(const int4*)(kr0 + 16 + 8 * hh + 4)));
      s0 = __builtin_amdgcn_wmma_f32_16x16x32_bf16(false, qf0, false, k00,
                                                   (short)0, s0, false, false);
      s0 = __builtin_amdgcn_wmma_f32_16x16x32_bf16(false, qf1, false, k01,
                                                   (short)0, s0, false, false);
      const int* kr1 = (const int*)(Kp + base + (size_t)(kb + 16 + nl) * 1024);
      v16bf k10 = as_bf(pack8(*(const int4*)(kr1 + 8 * hh),
                              *(const int4*)(kr1 + 8 * hh + 4)));
      v16bf k11 = as_bf(pack8(*(const int4*)(kr1 + 16 + 8 * hh),
                              *(const int4*)(kr1 + 16 + 8 * hh + 4)));
      s1 = __builtin_amdgcn_wmma_f32_16x16x32_bf16(false, qf0, false, k10,
                                                   (short)0, s1, false, false);
      s1 = __builtin_amdgcn_wmma_f32_16x16x32_bf16(false, qf1, false, k11,
                                                   (short)0, s1, false, false);
    }
    // ---- online softmax (row r = q0 + 8*hh + j lives in element j)
    float al[8];
    #pragma unroll
    for (int j = 0; j < 8; j++) {
      int r = q0 + 8 * hh + j;
      float a0 = s0[j] * 0.125f;
      float a1 = s1[j] * 0.125f;
      if (kb + nl > r)      a0 = -INFINITY;
      if (kb + 16 + nl > r) a1 = -INFINITY;
      float mx = redmax16(fmaxf(a0, a1));
      float nm = fmaxf(m8[j], mx);
      al[j] = __expf(m8[j] - nm);
      float p0 = __expf(a0 - nm);
      float p1 = __expf(a1 - nm);
      float rs = redsum16(p0 + p1);
      l8[j] = l8[j] * al[j] + rs;
      m8[j] = nm;
      int rr = (8 * hh + j) * 40;
      pPu[rr + nl]      = f2bf(p0);
      pPu[rr + 16 + nl] = f2bf(p1);
    }
    #pragma unroll
    for (int g = 0; g < 4; g++)
      #pragma unroll
      for (int j = 0; j < 8; j++) O[g][j] *= al[j];

    // ---- stage V^T (32 keys x 64 d -> Vt[d][key]) into per-wave LDS
    #pragma unroll
    for (int it = 0; it < 8; it++) {
      int idx = it * 32 + lane;      // 0..255
      int kk = idx >> 3, seg = idx & 7;
      const int* vp = (const int*)(Vp + base + (size_t)(kb + kk) * 1024) + seg * 4;
      int4 vv = *(const int4*)vp;
      int vr[4] = {vv.x, vv.y, vv.z, vv.w};
      #pragma unroll
      for (int e = 0; e < 4; e++) {
        int d0 = seg * 8 + 2 * e;
        u32 wv = (u32)vr[e];
        pVu[d0 * 40 + kk]       = (u16)(wv & 0xffffu);
        pVu[(d0 + 1) * 40 + kk] = (u16)(wv >> 16);
      }
    }

    // ---- O += P V  (A-frag of P from LDS, B-frags of V^T from LDS)
    const int* pr = pPw + nl * 20;
    v16bf pf = as_bf(pack8(*(const int4*)(pr + 8 * hh),
                           *(const int4*)(pr + 8 * hh + 4)));
    #pragma unroll
    for (int g = 0; g < 4; g++) {
      const int* vr2 = pVw + (g * 16 + nl) * 20 + 8 * hh;
      v16bf vf = as_bf(pack8(*(const int4*)vr2, *(const int4*)(vr2 + 4)));
      O[g] = __builtin_amdgcn_wmma_f32_16x16x32_bf16(false, pf, false, vf,
                                                     (short)0, O[g], false, false);
    }
  }

  float linv[8];
  #pragma unroll
  for (int j = 0; j < 8; j++) linv[j] = 1.0f / l8[j];
  #pragma unroll
  for (int g = 0; g < 4; g++)
    #pragma unroll
    for (int j = 0; j < 8; j++) {
      int r = q0 + 8 * hh + j;
      Y[base + (size_t)r * 1024 + g * 16 + nl] = f2bf(O[g][j] * linv[j]);
    }
}

// ---------------------------- host orchestration ----------------------------
extern "C" void kernel_launch(void* const* d_in, const int* in_sizes, int n_in,
                              void* d_out, int out_size, void* d_ws, size_t ws_size,
                              hipStream_t stream) {
  (void)in_sizes; (void)n_in; (void)out_size; (void)ws_size;
  const float* x     = (const float*)d_in[0];
  const float* Wemb1 = (const float*)d_in[1];
  const float* bemb1 = (const float*)d_in[2];
  const float* Wemb2 = (const float*)d_in[3];
  const float* bemb2 = (const float*)d_in[4];
  const float* ln1_g = (const float*)d_in[5];
  const float* ln1_b = (const float*)d_in[6];
  const float* Wq    = (const float*)d_in[7];
  const float* bq    = (const float*)d_in[8];
  const float* Wk    = (const float*)d_in[9];
  const float* bk    = (const float*)d_in[10];
  const float* Wv    = (const float*)d_in[11];
  const float* bv    = (const float*)d_in[12];
  const float* Wo    = (const float*)d_in[13];
  const float* bo    = (const float*)d_in[14];
  const float* ln2_g = (const float*)d_in[15];
  const float* ln2_b = (const float*)d_in[16];
  const float* W1    = (const float*)d_in[17];
  const float* b1    = (const float*)d_in[18];
  const float* W2    = (const float*)d_in[19];
  const float* b2    = (const float*)d_in[20];
  const float* lnf_g = (const float*)d_in[21];
  const float* lnf_b = (const float*)d_in[22];
  const float* Whead = (const float*)d_in[23];

  const int E = 1024, L = 12, M = 16 * 512, F = 4096;
  const size_t EE = (size_t)E * E, EF = (size_t)E * F;

  char* ws = (char*)d_ws;
  auto carve = [&](size_t bytes) -> char* {
    char* p = ws; ws += (bytes + 255) & ~(size_t)255; return p;
  };
  u16* Wemb2t = (u16*)carve(EE * 2);
  u16* Wqt    = (u16*)carve(L * EE * 2);
  u16* Wkt    = (u16*)carve(L * EE * 2);
  u16* Wvt    = (u16*)carve(L * EE * 2);
  u16* Wot    = (u16*)carve(L * EE * 2);
  u16* W1t    = (u16*)carve(L * EF * 2);
  u16* W2t    = (u16*)carve(L * EF * 2);
  u16* Wht    = (u16*)carve(EE * 2);
  float* hbuf = (float*)carve((size_t)M * E * 4);
  u16* abf    = (u16*)carve((size_t)M * E * 2);
  u16* qb     = (u16*)carve((size_t)M * E * 2);
  u16* kb     = (u16*)carve((size_t)M * E * 2);
  u16* vb     = (u16*)carve((size_t)M * E * 2);
  u16* yb     = (u16*)carve((size_t)M * E * 2);
  u16* ub     = (u16*)carve((size_t)M * F * 2);

  dim3 tb(32, 8);
  transpose_bf16<<<dim3(E / 32, E / 32, 1), tb, 0, stream>>>(Wemb2, Wemb2t, E, E);
  transpose_bf16<<<dim3(E / 32, E / 32, L), tb, 0, stream>>>(Wq, Wqt, E, E);
  transpose_bf16<<<dim3(E / 32, E / 32, L), tb, 0, stream>>>(Wk, Wkt, E, E);
  transpose_bf16<<<dim3(E / 32, E / 32, L), tb, 0, stream>>>(Wv, Wvt, E, E);
  transpose_bf16<<<dim3(E / 32, E / 32, L), tb, 0, stream>>>(Wo, Wot, E, E);
  transpose_bf16<<<dim3(F / 32, E / 32, L), tb, 0, stream>>>(W1, W1t, E, F);
  transpose_bf16<<<dim3(E / 32, F / 32, L), tb, 0, stream>>>(W2, W2t, F, E);
  transpose_bf16<<<dim3(E / 32, E / 32, 1), tb, 0, stream>>>(Whead, Wht, E, E);

  embed1_kernel<<<dim3(E / 256, M), 256, 0, stream>>>(x, Wemb1, bemb1, abf);
  gemm_bf16<1><<<dim3(M / 128, E / 128), 256, 0, stream>>>(
      abf, Wemb2t, bemb2, nullptr, hbuf, nullptr, M, E, E);

  for (int l = 0; l < L; l++) {
    ln_bf16<<<M, 256, 0, stream>>>(hbuf, ln1_g + l * E, ln1_b + l * E, abf);
    gemm_bf16<0><<<dim3(M / 128, E / 128), 256, 0, stream>>>(
        abf, Wqt + l * EE, bq + l * E, nullptr, nullptr, qb, M, E, E);
    gemm_bf16<0><<<dim3(M / 128, E / 128), 256, 0, stream>>>(
        abf, Wkt + l * EE, bk + l * E, nullptr, nullptr, kb, M, E, E);
    gemm_bf16<0><<<dim3(M / 128, E / 128), 256, 0, stream>>>(
        abf, Wvt + l * EE, bv + l * E, nullptr, nullptr, vb, M, E, E);
    attn_wmma<<<16 * 16 * 4, 256, 0, stream>>>(qb, kb, vb, yb);
    gemm_bf16<2><<<dim3(M / 128, E / 128), 256, 0, stream>>>(
        yb, Wot + l * EE, bo + l * E, hbuf, hbuf, nullptr, M, E, E);
    ln_bf16<<<M, 256, 0, stream>>>(hbuf, ln2_g + l * E, ln2_b + l * E, abf);
    gemm_bf16<3><<<dim3(M / 128, F / 128), 256, 0, stream>>>(
        abf, W1t + l * EF, b1 + l * F, nullptr, nullptr, ub, M, F, E);
    gemm_bf16<2><<<dim3(M / 128, E / 128), 256, 0, stream>>>(
        ub, W2t + l * EF, b2 + l * E, hbuf, hbuf, nullptr, M, E, F);
  }

  ln_bf16<<<M, 256, 0, stream>>>(hbuf, lnf_g, lnf_b, abf);
  gemm_bf16<4><<<dim3(M / 128, E / 128), 256, 0, stream>>>(
      abf, Wht, nullptr, nullptr, (float*)d_out, nullptr, M, E, E);
}
